// clip_scl_modified_2070174237307
// MI455X (gfx1250) — matrix operand order repaired and checked by reading, not compiled, not run on hardware
//
#include <hip/hip_runtime.h>

typedef __attribute__((ext_vector_type(16))) _Float16 v16h;
typedef __attribute__((ext_vector_type(8)))  float    v8f;

#define B_ 16
#define L_ 2048
#define D_ 512
#define C_GAUSS    1.2533141373155003f   // sqrt(2*pi)/2
#define LOGC_GAUSS 0.22579135264472733f  // log(C_GAUSS)

// ---------------------------------------------------------------------------
// Kernel 1: reciprocal L2 norms of each row of embs and clip_embs.
// One wave (32 lanes) per row; each lane handles 16 contiguous floats.
// ---------------------------------------------------------------------------
__global__ void norms_kernel(const float* __restrict__ embs,
                             const float* __restrict__ clip,
                             float* __restrict__ rnP, float* __restrict__ rnQ) {
    int wave = threadIdx.x >> 5, lane = threadIdx.x & 31;
    int row  = blockIdx.x * 8 + wave;
    const float4* pe = (const float4*)(embs + (size_t)row * D_) + lane * 4;
    const float4* qe = (const float4*)(clip + (size_t)row * D_) + lane * 4;
    float sp = 0.f, sq = 0.f;
#pragma unroll
    for (int k = 0; k < 4; ++k) {
        float4 a = pe[k]; sp += a.x*a.x + a.y*a.y + a.z*a.z + a.w*a.w;
        float4 b = qe[k]; sq += b.x*b.x + b.y*b.y + b.z*b.z + b.w*b.w;
    }
#pragma unroll
    for (int m = 16; m >= 1; m >>= 1) {
        sp += __shfl_xor(sp, m, 32);
        sq += __shfl_xor(sq, m, 32);
    }
    if (lane == 0) {
        rnP[row] = 1.0f / fmaxf(sqrtf(sp), 1e-12f);
        rnQ[row] = 1.0f / fmaxf(sqrtf(sq), 1e-12f);
    }
}

// ---------------------------------------------------------------------------
// Kernel 2: m[b] = count of mask>0 per batch row.
// ---------------------------------------------------------------------------
__global__ void mask_kernel(const int* __restrict__ mask, int* __restrict__ mArr) {
    __shared__ int red[256];
    int b = blockIdx.x, t = threadIdx.x;
    int s = 0;
    for (int i = t; i < L_; i += 256) s += (mask[b * L_ + i] > 0) ? 1 : 0;
    red[t] = s; __syncthreads();
    for (int o = 128; o > 0; o >>= 1) { if (t < o) red[t] += red[t + o]; __syncthreads(); }
    if (t == 0) mArr[b] = red[0];
}

// ---------------------------------------------------------------------------
// Kernel 3a: pack normalized f16 A-operand fragments (16x32, WMMA layout).
// ISA layout: lanes 0-15 row M=lane, halves 0..7 -> K=0..7, 8..15 -> K=16..23;
// lanes 16-31 same rows, K=8..15 and K=24..31.  => k = h + 8g + 8*(h>>3).
// Fragment index = ((b*128 + it)*16 + kc); each fragment = 32 lanes * 32B.
// ---------------------------------------------------------------------------
__global__ void packA_kernel(const float* __restrict__ embs,
                             const float* __restrict__ rn,
                             v16h* __restrict__ out) {
    unsigned tid = blockIdx.x * 256u + threadIdx.x;
    int lane = tid & 31;
    int kc   = (tid >> 5) & 15;
    int it   = (tid >> 9) & 127;
    int b    = tid >> 16;
    int g    = lane >> 4;
    int row  = it * 16 + (lane & 15);
    size_t base = (size_t)(b * L_ + row) * D_ + kc * 32 + g * 8;
    float s = rn[b * L_ + row];
    const float4* p0 = (const float4*)(embs + base);       // K run [8g, 8g+8)
    const float4* p1 = (const float4*)(embs + base + 16);  // K run [8g+16, 8g+24)
    float4 r0 = p0[0], r1 = p0[1], r2 = p1[0], r3 = p1[1];
    v16h o;
    o[0]=(_Float16)(r0.x*s); o[1]=(_Float16)(r0.y*s); o[2]=(_Float16)(r0.z*s); o[3]=(_Float16)(r0.w*s);
    o[4]=(_Float16)(r1.x*s); o[5]=(_Float16)(r1.y*s); o[6]=(_Float16)(r1.z*s); o[7]=(_Float16)(r1.w*s);
    o[8]=(_Float16)(r2.x*s); o[9]=(_Float16)(r2.y*s); o[10]=(_Float16)(r2.z*s); o[11]=(_Float16)(r2.w*s);
    o[12]=(_Float16)(r3.x*s); o[13]=(_Float16)(r3.y*s); o[14]=(_Float16)(r3.z*s); o[15]=(_Float16)(r3.w*s);
    out[tid] = o;
}

// ---------------------------------------------------------------------------
// Kernel 3b: pack normalized f16 B-operand fragments (32x16, WMMA layout).
// lanes 0-15 col N=lane hold K=0..7,16..23; lanes 16-31 hold K=8..15,24..31.
// ---------------------------------------------------------------------------
__global__ void packB_kernel(const float* __restrict__ clip,
                             const float* __restrict__ rn,
                             v16h* __restrict__ out) {
    unsigned tid = blockIdx.x * 256u + threadIdx.x;
    int lane = tid & 31;
    int kc   = (tid >> 5) & 15;
    int jt   = (tid >> 9) & 127;
    int b    = tid >> 16;
    int g    = lane >> 4;
    int col  = jt * 16 + (lane & 15);
    size_t base = (size_t)(b * L_ + col) * D_ + kc * 32 + g * 8;
    float s = rn[b * L_ + col];
    const float4* p = (const float4*)(clip + base);  // K runs [8g, 8g+8) and [8g+16, 8g+24)
    float4 r0 = p[0], r1 = p[1], r2 = p[4], r3 = p[5];
    v16h o;
    o[0]=(_Float16)(r0.x*s); o[1]=(_Float16)(r0.y*s); o[2]=(_Float16)(r0.z*s); o[3]=(_Float16)(r0.w*s);
    o[4]=(_Float16)(r1.x*s); o[5]=(_Float16)(r1.y*s); o[6]=(_Float16)(r1.z*s); o[7]=(_Float16)(r1.w*s);
    o[8]=(_Float16)(r2.x*s); o[9]=(_Float16)(r2.y*s); o[10]=(_Float16)(r2.z*s); o[11]=(_Float16)(r2.w*s);
    o[12]=(_Float16)(r3.x*s); o[13]=(_Float16)(r3.y*s); o[14]=(_Float16)(r3.z*s); o[15]=(_Float16)(r3.w*s);
    out[tid] = o;
}

// ---------------------------------------------------------------------------
// Kernel 4: WMMA GEMM with fused streaming softmax/KL reduction.
// Grid = B * (L/128) blocks; 8 waves, each wave owns a 16-row i-strip.
// The B fragments this block consumes are contiguous: fragment s = jt*16+kc
// lives at Qn + (b*2048 + s)*32 v16h.  A 4-deep rotating register buffer
// prefetches fragments so ~4 b128-load pairs stay in flight under the
// v_wmma chain (instead of s_wait_loadcnt 0x0 before every wmma).
// Qn is padded by 4 fragments so the tail prefetch can run off the end.
// ---------------------------------------------------------------------------
__global__ void __launch_bounds__(256)
gemm_kl_kernel(const v16h* __restrict__ Pn, const v16h* __restrict__ Qn,
               const int* __restrict__ mArr, float* __restrict__ rowsum) {
    int b    = blockIdx.x >> 4;
    int mt   = blockIdx.x & 15;
    int wave = threadIdx.x >> 5, lane = threadIdx.x & 31;
    int it   = mt * 8 + wave;
    int g    = lane >> 4, nl = lane & 15;
    int mb   = mArr[b];
    int ibase = it * 16 + g * 8;   // global row of accumulator slot r=0

    // Preload the wave's 16 A fragments (16 rows x full K=512) into registers.
    const v16h* Ap = Pn + (size_t)(b * 128 + it) * 16 * 32 + lane;
    v16h a[16];
#pragma unroll
    for (int kc = 0; kc < 16; ++kc) a[kc] = Ap[kc * 32];

    // Flat B-fragment stream for this block (fragment s at Bq[s*32]).
    const v16h* Bq = Qn + (size_t)b * 2048 * 32 + lane;

    float acc1[8], acc2[8];
#pragma unroll
    for (int r = 0; r < 8; ++r) { acc1[r] = 0.f; acc2[r] = 0.f; }

    // Prime the 4-deep prefetch pipeline.
    v16h buf[4];
#pragma unroll
    for (int i = 0; i < 4; ++i) buf[i] = Bq[i * 32];

    for (int jt = 0; jt < 128; ++jt) {
        const v16h* Bjt = Bq + (size_t)jt * 16 * 32;
        v8f c = {0.f, 0.f, 0.f, 0.f, 0.f, 0.f, 0.f, 0.f};
#pragma unroll
        for (int kc = 0; kc < 16; ++kc) {
            v16h cur = buf[kc & 3];
            buf[kc & 3] = Bjt[(kc + 4) * 32];   // prefetch fragment s+4
            c = __builtin_amdgcn_wmma_f32_16x16x32_f16(
                    false, a[kc], false, cur, (short)0, c, false, false);
        }
        int n = jt * 16 + nl;
        bool cv = n < mb;
#pragma unroll
        for (int r = 0; r < 8; ++r) {
            float lg = c[r];
            acc1[r] += cv ? __expf(lg) : 0.0f;
            float d = (float)(ibase + r - n);
            float w = (cv && fabsf(d) < 32.0f)
                          ? (C_GAUSS * __expf(-0.125f * d * d)) : 0.0f;
            acc2[r] = __builtin_fmaf(w, lg, acc2[r]);
        }
    }

    // Reduce across the 16 lanes that share each row (half-wave xor reduce).
#pragma unroll
    for (int r = 0; r < 8; ++r) {
#pragma unroll
        for (int msk = 8; msk >= 1; msk >>= 1) {
            acc1[r] += __shfl_xor(acc1[r], msk, 16);
            acc2[r] += __shfl_xor(acc2[r], msk, 16);
        }
    }

    if (nl == 0) {
        for (int r = 0; r < 8; ++r) {
            int i = ibase + r;
            // Analytic band pass for Zt = sum w and S3 = sum w*log(w); w
            // underflows for |i-j| > 31 so the band is exact in f32.
            float Zt = 0.f, S3 = 0.f;
            int j0 = i - 31; if (j0 < 0) j0 = 0;
            int j1 = i + 31; if (j1 > mb - 1) j1 = mb - 1;
            for (int j = j0; j <= j1; ++j) {
                float d = (float)(i - j);
                float q = -0.125f * d * d;
                float w = C_GAUSS * __expf(q);
                Zt += w;
                S3 = __builtin_fmaf(w, LOGC_GAUSS + q, S3);
            }
            // row_sum = LSE + (S3 - acc2)/Zt - log Zt
            float rs = __logf(acc1[r]) + (S3 - acc2[r]) / Zt - __logf(Zt);
            rowsum[b * L_ + i] = rs;
        }
    }
}

// ---------------------------------------------------------------------------
// Kernel 5: final masked mean over rows and batches -> scalar.
// ---------------------------------------------------------------------------
__global__ void final_kernel(const float* __restrict__ rowsum,
                             const int* __restrict__ mArr,
                             float* __restrict__ out) {
    __shared__ float red[256];
    int t = threadIdx.x;
    float acc = 0.f;
    for (int b = 0; b < B_; ++b) {
        int mb = mArr[b];
        float inv = 1.0f / (float)mb;
        for (int i = t; i < mb; i += 256) acc += rowsum[b * L_ + i] * inv;
    }
    red[t] = acc; __syncthreads();
    for (int o = 128; o > 0; o >>= 1) { if (t < o) red[t] += red[t + o]; __syncthreads(); }
    if (t == 0) out[0] = red[0] / (float)B_;
}

// ---------------------------------------------------------------------------
// Workspace layout (bytes):
//   [0,256)                 : m[16] ints
//   [256, +128K)            : rnP
//   [+128K, +256K)          : rnQ
//   [+256K, +384K)          : rowsum
//   [393472, +32M)          : packed f16 P fragments
//   [33947904, +32M+4K)     : packed f16 Q fragments (+4 fragments pad for
//                             the prefetch pipeline tail)
// Total ~64.4 MB.
// ---------------------------------------------------------------------------
extern "C" void kernel_launch(void* const* d_in, const int* in_sizes, int n_in,
                              void* d_out, int out_size, void* d_ws, size_t ws_size,
                              hipStream_t stream) {
    const float* embs = (const float*)d_in[0];
    const float* clip = (const float*)d_in[1];
    const int*   mask = (const int*)d_in[2];

    char*  ws     = (char*)d_ws;
    int*   mArr   = (int*)ws;
    float* rnP    = (float*)(ws + 256);
    float* rnQ    = (float*)(ws + 256 + 131072);
    float* rowsum = (float*)(ws + 256 + 2 * 131072);
    v16h*  Pn     = (v16h*)(ws + 393472);
    v16h*  Qn     = (v16h*)(ws + 393472 + 33554432);

    norms_kernel<<<(B_ * L_) / 8, 256, 0, stream>>>(embs, clip, rnP, rnQ);
    mask_kernel <<<B_, 256, 0, stream>>>(mask, mArr);
    packA_kernel<<<4096, 256, 0, stream>>>(embs, rnP, Pn);
    packB_kernel<<<4096, 256, 0, stream>>>(clip, rnQ, Qn);
    gemm_kl_kernel<<<B_ * (L_ / 128), 256, 0, stream>>>(Pn, Qn, mArr, rowsum);
    final_kernel<<<1, 256, 0, stream>>>(rowsum, mArr, (float*)d_out);
}
